// MultiHeadAttention_32693291057906
// MI455X (gfx1250) — matrix operand order repaired and checked by reading, not compiled, not run on hardware
//
#include <hip/hip_runtime.h>

// ---------------------------------------------------------------------------
// MHA for MI455X (gfx1250, wave32). All matmuls via v_wmma_f32_16x16x32_bf16.
// LDS staging via global_load_async_to_lds_b128 (ASYNCcnt), double-buffered.
// B=4, S=2048, D=1024, H=16, DK=64.
// ---------------------------------------------------------------------------

#define BB   4
#define SS   2048
#define DD   1024
#define HH   16
#define DKK  64
#define SW   (SS / 32)   // 64 mask words per row

typedef __bf16 bf16_t;
typedef __attribute__((ext_vector_type(16))) __bf16 v16bf;
typedef __attribute__((ext_vector_type(8)))  __bf16 v8bf;
typedef __attribute__((ext_vector_type(8)))  float  v8f;

union FragAB { v16bf full; v8bf half[2]; };

// xor-butterfly over lanes via ds_swizzle (group-of-32: and=0x1f, or=0, xor=m)
#define SWZ_XOR_F(x, m) \
  __int_as_float(__builtin_amdgcn_ds_swizzle(__float_as_int(x), (((m) << 10) | 0x1f)))

// memory -> LDS DMA (per-lane 16B), tracked by ASYNCcnt. INST_OFFSET is added
// to BOTH the LDS and the global address (CDNA5 ISA).
#define ASYNC_B128(ldsoff, gaddr)                                           \
  asm volatile("global_load_async_to_lds_b128 %0, %1, off" ::"v"(ldsoff),   \
               "v"(gaddr)                                                   \
               : "memory")
#define ASYNC_B128_O16(ldsoff, gaddr)                                       \
  asm volatile("global_load_async_to_lds_b128 %0, %1, off offset:16" ::"v"( \
                   ldsoff),                                                 \
               "v"(gaddr)                                                   \
               : "memory")
#define WAIT_ASYNC0() asm volatile("s_wait_asynccnt 0x0" ::: "memory")
#define WAIT_DS0()    asm volatile("s_wait_dscnt 0x0" ::: "memory")

__device__ __forceinline__ unsigned lds_off(const void* p) {
  return (unsigned)(uintptr_t)p;  // low 32 bits of shared aperture = LDS byte addr
}

// ---------------------------------------------------------------------------
// f32 -> bf16, 4 elements / thread
// ---------------------------------------------------------------------------
__global__ void cvt_kernel(const float* __restrict__ src, bf16_t* __restrict__ dst,
                           size_t n4) {
  size_t i = (size_t)blockIdx.x * blockDim.x + threadIdx.x;
  if (i < n4) {
    float4 f = ((const float4*)src)[i];
    union { bf16_t h[4]; uint2 u; } r;
    r.h[0] = (bf16_t)f.x; r.h[1] = (bf16_t)f.y;
    r.h[2] = (bf16_t)f.z; r.h[3] = (bf16_t)f.w;
    ((uint2*)dst)[i] = r.u;
  }
}

// ---------------------------------------------------------------------------
// int32 mask -> bitmask (1 bit per key). word w covers flat ints [w*32, w*32+32)
// ---------------------------------------------------------------------------
__global__ void maskpack_kernel(const int* __restrict__ mask,
                                unsigned int* __restrict__ out, int nwords) {
  int w = blockIdx.x * blockDim.x + threadIdx.x;
  if (w < nwords) {
    const int* p = mask + (size_t)w * 32;
    unsigned int bits = 0u;
#pragma unroll
    for (int i = 0; i < 32; ++i) bits |= (p[i] != 0) ? (1u << i) : 0u;
    out[w] = bits;
  }
}

// ---------------------------------------------------------------------------
// GEMM:  Y[m,n] = sum_k A[m,k] * W[n,k] + bias[n]   (torch Linear: x @ W.T + b)
// A: [M,K] bf16 row-major,  W: [N,K] bf16 row-major.
// Block 256 thr (8 waves), tile 128x128, K-step 32, async double-buffered LDS.
// MODE 0: bf16 [M,N] row-major.  MODE 1: f32 [M,N].  MODE 2: bf16 [B,H,DK,S].
// ---------------------------------------------------------------------------
template <int MODE>
__global__ __launch_bounds__(256)
void gemm_kernel(const bf16_t* __restrict__ A, const bf16_t* __restrict__ W,
                 const float* __restrict__ bias, void* __restrict__ out,
                 int M, int N, int K) {
  __shared__ bf16_t As[2][128 * 32];
  __shared__ bf16_t Bs[2][128 * 32];

  const int tid = threadIdx.x;
  const int wid = tid >> 5, lane = tid & 31;
  const int l16 = lane & 15, hi = lane >> 4;
  const int wm = wid & 3, wn = wid >> 2;           // 4x2 wave grid
  const int m0 = blockIdx.y * 128, n0 = blockIdx.x * 128;

  v8f acc[2][4] = {};

  // each thread stages one contiguous 32B run: row lr, cols lc..lc+15
  const int lr = tid >> 1, lc = (tid & 1) * 16;
  const unsigned ldsA0 = lds_off(&As[0][lr * 32 + lc]);
  const unsigned ldsB0 = lds_off(&Bs[0][lr * 32 + lc]);
  const unsigned bufB = (unsigned)(128 * 32 * 2);  // bytes between buffers
  const unsigned long long gA =
      (unsigned long long)(uintptr_t)(A + (size_t)(m0 + lr) * K + lc);
  const unsigned long long gB =
      (unsigned long long)(uintptr_t)(W + (size_t)(n0 + lr) * K + lc);

  auto issue = [&](int buf, int k0) {
    unsigned la = ldsA0 + (unsigned)buf * bufB;
    unsigned lb = ldsB0 + (unsigned)buf * bufB;
    unsigned long long a0 = gA + (unsigned long long)k0 * 2;
    unsigned long long b0 = gB + (unsigned long long)k0 * 2;
    ASYNC_B128(la, a0);
    ASYNC_B128_O16(la, a0);
    ASYNC_B128(lb, b0);
    ASYNC_B128_O16(lb, b0);
  };

  const int nsteps = K / 32;
  issue(0, 0);

  for (int s = 0; s < nsteps; ++s) {
    const int buf = s & 1;
    WAIT_ASYNC0();        // this wave's DMA for buf has landed in LDS
    __syncthreads();      // everyone's DMA landed; everyone done reading buf^1
    if (s + 1 < nsteps) issue(buf ^ 1, (s + 1) * 32);

    FragAB af[2], bf[4];
#pragma unroll
    for (int i = 0; i < 2; ++i) {
      // A frag: lane row = l16; lo lanes k {0..7,16..23}, hi lanes +8
      const bf16_t* p = &As[buf][(wm * 32 + i * 16 + l16) * 32 + (hi ? 8 : 0)];
      af[i].half[0] = *(const v8bf*)p;
      af[i].half[1] = *(const v8bf*)(p + 16);
    }
#pragma unroll
    for (int jf = 0; jf < 4; ++jf) {
      // B frag: lane col = l16; lo lanes k {0..7,16..23}, hi lanes +8
      const bf16_t* p = &Bs[buf][(wn * 64 + jf * 16 + l16) * 32 + (hi ? 8 : 0)];
      bf[jf].half[0] = *(const v8bf*)p;
      bf[jf].half[1] = *(const v8bf*)(p + 16);
    }
#pragma unroll
    for (int i = 0; i < 2; ++i)
#pragma unroll
      for (int jf = 0; jf < 4; ++jf)
        acc[i][jf] = __builtin_amdgcn_wmma_f32_16x16x32_bf16(
            false, af[i].full, false, bf[jf].full, (short)0, acc[i][jf],
            false, false);
  }

  // epilogue: C layout -> lane element v is (row v + 8*hi, col l16)
#pragma unroll
  for (int i = 0; i < 2; ++i)
#pragma unroll
    for (int jf = 0; jf < 4; ++jf)
#pragma unroll
      for (int v = 0; v < 8; ++v) {
        int m = m0 + wm * 32 + i * 16 + v + 8 * hi;
        int n = n0 + wn * 64 + jf * 16 + l16;
        float val = acc[i][jf][v] + bias[n];
        if (MODE == 0) {
          ((bf16_t*)out)[(size_t)m * N + n] = (bf16_t)val;
        } else if (MODE == 1) {
          ((float*)out)[(size_t)m * N + n] = val;
        } else {
          int bb = m / SS, s = m % SS;
          int hh = n >> 6, dk = n & 63;
          ((bf16_t*)out)[((size_t)(bb * HH + hh) * DKK + dk) * SS + s] =
              (bf16_t)val;
        }
      }
}

// ---------------------------------------------------------------------------
// Flash attention. Block = 256 thr = 8 waves; wave owns 16 q rows.
// Qb,Kb: bf16 [B,S,D]; Vt: bf16 [B,H,DK,S]; mbits: 1 bit/key; ctx: bf16 [B,S,D].
// K/V tiles async-DMA'd into double-buffered LDS.
// ---------------------------------------------------------------------------
__global__ __launch_bounds__(256)
void attn_kernel(const bf16_t* __restrict__ Qb, const bf16_t* __restrict__ Kb,
                 const bf16_t* __restrict__ Vt,
                 const unsigned int* __restrict__ mbits,
                 bf16_t* __restrict__ ctx) {
  __shared__ bf16_t Ks[2][32 * 64];  // [key][d]
  __shared__ bf16_t Vs[2][64 * 32];  // [dk][key]
  __shared__ bf16_t Pt[8][16 * 32];  // per-wave P staging

  const int tid = threadIdx.x;
  const int wid = tid >> 5, lane = tid & 31;
  const int l16 = lane & 15, hi = lane >> 4;
  const int bh = blockIdx.y, b = bh / HH, h = bh % HH;
  const int qbase = blockIdx.x * 128 + wid * 16;

  // Q A-fragments for the two 32-wide d chunks (loaded once)
  FragAB qa[2];
  {
    const bf16_t* qp =
        Qb + (size_t)(b * SS + qbase + l16) * DD + h * DKK + (hi ? 8 : 0);
#pragma unroll
    for (int d0 = 0; d0 < 2; ++d0) {
      qa[d0].half[0] = *(const v8bf*)(qp + d0 * 32);
      qa[d0].half[1] = *(const v8bf*)(qp + d0 * 32 + 16);
    }
  }

  // async staging addresses: each thread moves one 16B run of K and of V
  const int kr = tid >> 3, kc = (tid & 7) * 8;   // K tile 32x64
  const int vr = tid >> 2, vc = (tid & 3) * 8;   // V tile 64x32
  const unsigned ldsK0 = lds_off(&Ks[0][kr * 64 + kc]);
  const unsigned ldsV0 = lds_off(&Vs[0][vr * 32 + vc]);
  const unsigned long long gK0 =
      (unsigned long long)(uintptr_t)(Kb + (size_t)(b * SS + kr) * DD +
                                      h * DKK + kc);
  const unsigned long long gV0 =
      (unsigned long long)(uintptr_t)(Vt + ((size_t)bh * DKK + vr) * SS + vc);

  auto issueKV = [&](int buf, int j) {
    unsigned lk = ldsK0 + (unsigned)buf * 4096u;
    unsigned lv = ldsV0 + (unsigned)buf * 4096u;
    unsigned long long gk = gK0 + (unsigned long long)j * (32ull * DD * 2);
    unsigned long long gv = gV0 + (unsigned long long)j * 64ull;  // 32 keys * 2B
    ASYNC_B128(lk, gk);
    ASYNC_B128(lv, gv);
  };

  float mrow[8], lrow[8];
  v8f o[4] = {};
#pragma unroll
  for (int v = 0; v < 8; ++v) { mrow[v] = -3.0e38f; lrow[v] = 0.f; }

  const int nj = SS / 32;
  issueKV(0, 0);

  for (int j = 0; j < nj; ++j) {
    const int buf = j & 1;
    WAIT_ASYNC0();
    __syncthreads();
    if (j + 1 < nj) issueKV(buf ^ 1, j + 1);

    // scores: two 16(q)x16(key) tiles, reduce over DK=64 in two K=32 steps
    v8f st[2] = {};
#pragma unroll
    for (int t = 0; t < 2; ++t)
#pragma unroll
      for (int d0 = 0; d0 < 2; ++d0) {
        FragAB kf;
        const bf16_t* kp = &Ks[buf][(t * 16 + l16) * 64 + d0 * 32 + (hi ? 8 : 0)];
        kf.half[0] = *(const v8bf*)(kp);
        kf.half[1] = *(const v8bf*)(kp + 16);
        st[t] = __builtin_amdgcn_wmma_f32_16x16x32_bf16(
            false, qa[d0].full, false, kf.full, (short)0, st[t], false, false);
      }

    // masked online softmax; row stats per C-layout row (v + 8*hi)
    float p0a[8], p1a[8];
#pragma unroll
    for (int v = 0; v < 8; ++v) {
      const int q = qbase + v + 8 * hi;
      unsigned int w = mbits[(size_t)(b * SS + q) * SW + j];
      float s0 = ((w >> l16) & 1u)        ? st[0][v] * 0.125f : -1.0e9f;
      float s1 = ((w >> (16 + l16)) & 1u) ? st[1][v] * 0.125f : -1.0e9f;
      float mx = fmaxf(s0, s1);
      mx = fmaxf(mx, SWZ_XOR_F(mx, 1));
      mx = fmaxf(mx, SWZ_XOR_F(mx, 2));
      mx = fmaxf(mx, SWZ_XOR_F(mx, 4));
      mx = fmaxf(mx, SWZ_XOR_F(mx, 8));
      float nm = fmaxf(mrow[v], mx);
      float scl = __expf(mrow[v] - nm);
      float p0 = __expf(s0 - nm), p1 = __expf(s1 - nm);
      float rs = p0 + p1;
      rs += SWZ_XOR_F(rs, 1);
      rs += SWZ_XOR_F(rs, 2);
      rs += SWZ_XOR_F(rs, 4);
      rs += SWZ_XOR_F(rs, 8);
      lrow[v] = lrow[v] * scl + rs;
      mrow[v] = nm;
#pragma unroll
      for (int jj = 0; jj < 4; ++jj) o[jj][v] *= scl;
      p0a[v] = p0; p1a[v] = p1;
    }

    // stage P (16x32 bf16) into per-wave LDS region, reload as A-fragment
#pragma unroll
    for (int v = 0; v < 8; ++v) {
      Pt[wid][(v + 8 * hi) * 32 + l16]      = (bf16_t)p0a[v];
      Pt[wid][(v + 8 * hi) * 32 + 16 + l16] = (bf16_t)p1a[v];
    }
    WAIT_DS0();  // same-wave cross-lane LDS dependence

    FragAB pa;
    {
      const bf16_t* pp = &Pt[wid][l16 * 32 + (hi ? 8 : 0)];
      pa.half[0] = *(const v8bf*)(pp);
      pa.half[1] = *(const v8bf*)(pp + 16);
    }

    // O[16x64] += P[16x32] @ V[32x64]  (4 dk tiles)
#pragma unroll
    for (int jj = 0; jj < 4; ++jj) {
      FragAB vf;
      const bf16_t* vp = &Vs[buf][(jj * 16 + l16) * 32 + (hi ? 8 : 0)];
      vf.half[0] = *(const v8bf*)(vp);
      vf.half[1] = *(const v8bf*)(vp + 16);
      o[jj] = __builtin_amdgcn_wmma_f32_16x16x32_bf16(
          false, pa.full, false, vf.full, (short)0, o[jj], false, false);
    }
  }

  // epilogue: normalize, write ctx bf16 [B,S,D]
#pragma unroll
  for (int jj = 0; jj < 4; ++jj)
#pragma unroll
    for (int v = 0; v < 8; ++v) {
      int q = qbase + v + 8 * hi;
      int col = h * DKK + jj * 16 + l16;
      float val = o[jj][v] / lrow[v];
      ctx[(size_t)(b * SS + q) * DD + col] = (bf16_t)val;
    }
}

// ---------------------------------------------------------------------------
extern "C" void kernel_launch(void* const* d_in, const int* in_sizes, int n_in,
                              void* d_out, int out_size, void* d_ws,
                              size_t ws_size, hipStream_t stream) {
  (void)in_sizes; (void)n_in; (void)out_size; (void)ws_size;
  const float* q    = (const float*)d_in[0];
  const float* k    = (const float*)d_in[1];
  const float* v    = (const float*)d_in[2];
  const int*   mask = (const int*)d_in[3];
  const float* Wq = (const float*)d_in[4];  const float* bq = (const float*)d_in[5];
  const float* Wk = (const float*)d_in[6];  const float* bk = (const float*)d_in[7];
  const float* Wv = (const float*)d_in[8];  const float* bv = (const float*)d_in[9];
  const float* Wo = (const float*)d_in[10]; const float* bo = (const float*)d_in[11];
  float* out = (float*)d_out;

  const size_t nx = (size_t)BB * SS * DD;  // 8,388,608 activation elems
  const size_t nw = (size_t)DD * DD;       // 1,048,576 weight elems
  const size_t nmw = (size_t)BB * SS * SW; // 524,288 mask words

  char* ws = (char*)d_ws;
  size_t off = 0;
  auto alloc = [&](size_t bytes) {
    void* p = ws + off;
    off += (bytes + 255) & ~(size_t)255;
    return p;
  };
  bf16_t* xq  = (bf16_t*)alloc(nx * 2);
  bf16_t* xk  = (bf16_t*)alloc(nx * 2);
  bf16_t* xv  = (bf16_t*)alloc(nx * 2);
  bf16_t* wqb = (bf16_t*)alloc(nw * 2);
  bf16_t* wkb = (bf16_t*)alloc(nw * 2);
  bf16_t* wvb = (bf16_t*)alloc(nw * 2);
  bf16_t* wob = (bf16_t*)alloc(nw * 2);
  bf16_t* Qb  = (bf16_t*)alloc(nx * 2);
  bf16_t* Kb  = (bf16_t*)alloc(nx * 2);
  bf16_t* Vt  = (bf16_t*)alloc(nx * 2);
  unsigned int* mb = (unsigned int*)alloc(nmw * 4);
  bf16_t* ctx = xq;  // xq dead after Q projection; reuse for attention output

  const int M = BB * SS, N = DD, K = DD;

  // 1) precision conversion
  cvt_kernel<<<(unsigned)(nx / 4 / 256), 256, 0, stream>>>(q, xq, nx / 4);
  cvt_kernel<<<(unsigned)(nx / 4 / 256), 256, 0, stream>>>(k, xk, nx / 4);
  cvt_kernel<<<(unsigned)(nx / 4 / 256), 256, 0, stream>>>(v, xv, nx / 4);
  cvt_kernel<<<(unsigned)(nw / 4 / 256), 256, 0, stream>>>(Wq, wqb, nw / 4);
  cvt_kernel<<<(unsigned)(nw / 4 / 256), 256, 0, stream>>>(Wk, wkb, nw / 4);
  cvt_kernel<<<(unsigned)(nw / 4 / 256), 256, 0, stream>>>(Wv, wvb, nw / 4);
  cvt_kernel<<<(unsigned)(nw / 4 / 256), 256, 0, stream>>>(Wo, wob, nw / 4);

  // 2) mask -> bits
  maskpack_kernel<<<(unsigned)(nmw / 256), 256, 0, stream>>>(mask, mb, (int)nmw);

  // 3) projections
  dim3 ggrid(N / 128, M / 128);
  gemm_kernel<0><<<ggrid, 256, 0, stream>>>(xq, wqb, bq, (void*)Qb, M, N, K);
  gemm_kernel<0><<<ggrid, 256, 0, stream>>>(xk, wkb, bk, (void*)Kb, M, N, K);
  gemm_kernel<2><<<ggrid, 256, 0, stream>>>(xv, wvb, bv, (void*)Vt, M, N, K);

  // 4) flash attention
  attn_kernel<<<dim3(SS / 128, BB * HH), 256, 0, stream>>>(Qb, Kb, Vt, mb, ctx);

  // 5) output projection (f32 out)
  gemm_kernel<1><<<ggrid, 256, 0, stream>>>(ctx, wob, bo, (void*)out, M, N, K);
}
